// VMD_Network_38663295598643
// MI455X (gfx1250) — compile-verified, matrix operand order checked by reading, not run on hardware
//
#include <hip/hip_runtime.h>

typedef _Float16 f16;
typedef __attribute__((ext_vector_type(16))) _Float16 v16h;
typedef __attribute__((ext_vector_type(8)))  _Float16 v8h;
typedef __attribute__((ext_vector_type(8)))  float    v8f;

#define B_  8
#define C_  256
#define H_  128
#define W_  128
#define IN_ 64    // INTER
#define H2  130   // padded
#define W2  130   // padded

union U16 { v16h v; v8h h[2]; };

__device__ inline v16h frag_load(const f16* row, int hi) {
  U16 u;
  u.h[0] = *reinterpret_cast<const v8h*>(row + hi * 8);
  u.h[1] = *reinterpret_cast<const v8h*>(row + 16 + hi * 8);
  return u.v;
}
__device__ inline v8f acc_zero() {
  v8f r;
#pragma unroll
  for (int i = 0; i < 8; ++i) r[i] = 0.0f;
  return r;
}
__device__ inline float dot8(v8h a, v8h b) {
  float s = 0.f;
#pragma unroll
  for (int i = 0; i < 8; ++i) s += (float)a[i] * (float)b[i];
  return s;
}
#define WMMA(a, b, c) __builtin_amdgcn_wmma_f32_16x16x32_f16(false, (a), false, (b), (short)0, (c), false, false)

// ---------------- prep kernels ----------------

// NCHW f32 -> padded NHWC f16 [B][H2][W2][C] (borders zero)
__global__ __launch_bounds__(256) void k_cvt_pad(const float* __restrict__ src,
                                                 f16* __restrict__ dst, int total) {
  int t = blockIdx.x * 256 + threadIdx.x;
  if (t >= total) return;
  int c = t & 255;
  int rest = t >> 8;
  int wp = rest % W2;
  rest /= W2;
  int hp = rest % H2;
  int b = rest / H2;
  f16 v = (f16)0.0f;
  if (hp >= 1 && hp <= H_ && wp >= 1 && wp <= W_)
    v = (f16)src[((size_t)(b * C_ + c) * H_ + (hp - 1)) * W_ + (wp - 1)];
  dst[t] = v;
}

__global__ __launch_bounds__(256) void k_zero16(f16* __restrict__ p, int n) {
  int t = blockIdx.x * 256 + threadIdx.x;
  if (t < n) p[t] = (f16)0.0f;
}

// OIHW f32 -> [O][tap][Cin] f16
__global__ __launch_bounds__(256) void k_reorder_w(const float* __restrict__ src,
                                                   f16* __restrict__ dst,
                                                   int Cin, int total) {
  int t = blockIdx.x * 256 + threadIdx.x;
  if (t >= total) return;
  int kp = 9 * Cin;
  int o = t / kp;
  int rem = t - o * kp;
  int tap = rem / Cin;
  int c = rem - tap * Cin;
  dst[t] = (f16)src[((size_t)o * Cin + c) * 9 + tap];
}

// (8,64) f32 -> (16,64) f16 zero-padded rows 8..15
__global__ __launch_bounds__(256) void k_pad_qk(const float* __restrict__ src,
                                                f16* __restrict__ dst) {
  int t = blockIdx.x * 256 + threadIdx.x;  // t < 1024
  int o = t >> 6, c = t & 63;
  dst[t] = (o < 8) ? (f16)src[o * 64 + c] : (f16)0.0f;
}

__global__ __launch_bounds__(256) void k_cvt_small(const float* __restrict__ src,
                                                   f16* __restrict__ dst, int n) {
  int t = blockIdx.x * 256 + threadIdx.x;
  if (t < n) dst[t] = (f16)src[t];
}

// ---------------- conv kernels (implicit GEMM, WMMA, padded input) ----------------

// head conv: x padded NHWC(256) f16, w [64][9*256], out padded NHWC(64) f16, BN+ReLU
__global__ __launch_bounds__(256) void k_conv_head(const f16* __restrict__ x,
                                                   const f16* __restrict__ wA,
                                                   const float* __restrict__ sc,
                                                   const float* __restrict__ bi,
                                                   f16* __restrict__ y) {
  const int Cin = C_;
  const int bh = blockIdx.x;
  const int b = bh >> 7, h = bh & 127;
  const int wv = threadIdx.x >> 5, lane = threadIdx.x & 31;
  const int lr = lane & 15, hi = lane >> 4;
  const int m0 = (wv & 3) * 16;   // oc strip
  const int n0 = (wv >> 2) * 64;  // w half
  const int Kp = 9 * Cin;

  v8f acc[4];
#pragma unroll
  for (int j = 0; j < 4; ++j) acc[j] = acc_zero();

  const f16* arow = wA + (size_t)(m0 + lr) * Kp;

#pragma unroll
  for (int tap = 0; tap < 9; ++tap) {
    const int ty = tap / 3, tx = tap % 3;   // constants after unroll
    const f16* xrow = x + (size_t)(b * H2 + h + ty) * W2 * Cin;
#pragma unroll 2
    for (int c0 = 0; c0 < Cin; c0 += 32) {
      v16h a = frag_load(arow + tap * Cin + c0, hi);
      v16h bfr[4];
#pragma unroll
      for (int j = 0; j < 4; ++j)
        bfr[j] = frag_load(xrow + (size_t)(n0 + j * 16 + lr + tx) * Cin + c0, hi);
#pragma unroll
      for (int j = 0; j < 4; ++j)
        acc[j] = WMMA(a, bfr[j], acc[j]);
    }
  }
  float sv[8], bv[8];
#pragma unroll
  for (int r = 0; r < 8; ++r) {
    int m = m0 + hi * 8 + r;
    sv[r] = sc[m]; bv[r] = bi[m];
  }
#pragma unroll
  for (int j = 0; j < 4; ++j) {
    const int n = n0 + j * 16 + lr;
    f16* yp = y + ((size_t)(b * H2 + h + 1) * W2 + (n + 1)) * IN_;
#pragma unroll
    for (int r = 0; r < 8; ++r) {
      int m = m0 + hi * 8 + r;
      float v = acc[j][r] * sv[r] + bv[r];
      yp[m] = (f16)(v > 0.f ? v : 0.f);
    }
  }
}

// tail conv: x padded NHWC(64) f16, w [256][9*64], out f32 NCHW, BN+ReLU
__global__ __launch_bounds__(256) void k_conv_tail(const f16* __restrict__ x,
                                                   const f16* __restrict__ wA,
                                                   const float* __restrict__ sc,
                                                   const float* __restrict__ bi,
                                                   float* __restrict__ out) {
  const int Cin = IN_;
  const int bh = blockIdx.x;
  const int ocb = blockIdx.y;  // 0..3
  const int b = bh >> 7, h = bh & 127;
  const int wv = threadIdx.x >> 5, lane = threadIdx.x & 31;
  const int lr = lane & 15, hi = lane >> 4;
  const int m0 = (wv & 3) * 16;
  const int n0 = (wv >> 2) * 64;
  const int Kp = 9 * Cin;

  v8f acc[4];
#pragma unroll
  for (int j = 0; j < 4; ++j) acc[j] = acc_zero();

  const f16* arow = wA + (size_t)(ocb * 64 + m0 + lr) * Kp;

#pragma unroll
  for (int tap = 0; tap < 9; ++tap) {
    const int ty = tap / 3, tx = tap % 3;
    const f16* xrow = x + (size_t)(b * H2 + h + ty) * W2 * Cin;
#pragma unroll
    for (int c0 = 0; c0 < Cin; c0 += 32) {
      v16h a = frag_load(arow + tap * Cin + c0, hi);
      v16h bfr[4];
#pragma unroll
      for (int j = 0; j < 4; ++j)
        bfr[j] = frag_load(xrow + (size_t)(n0 + j * 16 + lr + tx) * Cin + c0, hi);
#pragma unroll
      for (int j = 0; j < 4; ++j)
        acc[j] = WMMA(a, bfr[j], acc[j]);
    }
  }
#pragma unroll
  for (int j = 0; j < 4; ++j) {
    const int n = n0 + j * 16 + lr;
#pragma unroll
    for (int r = 0; r < 8; ++r) {
      int mg = ocb * 64 + m0 + hi * 8 + r;
      float v = acc[j][r] * sc[mg] + bi[mg];
      out[((size_t)(b * C_ + mg) * H_ + h) * W_ + n] = (v > 0.f ? v : 0.f);
    }
  }
}

// ---------------- attention kernels ----------------

// q/k projection: O=8 padded to 16. in padded NHWC(64), out dense [b][h][w][8]
__global__ __launch_bounds__(256) void k_proj_qk(const f16* __restrict__ xin,
                                                 const f16* __restrict__ w16,
                                                 const float* __restrict__ bias,
                                                 f16* __restrict__ out) {
  const int bh = blockIdx.x;
  const int b = bh >> 7, h = bh & 127;
  const int wv = threadIdx.x >> 5, lane = threadIdx.x & 31;
  const int lr = lane & 15, hi = lane >> 4;
  const int n0 = wv * 16;

  v8f acc = acc_zero();
  const f16* xrow = xin + ((size_t)(b * H2 + h + 1) * W2 + 1) * IN_;
#pragma unroll
  for (int c0 = 0; c0 < IN_; c0 += 32) {
    v16h a = frag_load(w16 + lr * IN_ + c0, hi);
    v16h bf = frag_load(xrow + (size_t)(n0 + lr) * IN_ + c0, hi);
    acc = WMMA(a, bf, acc);
  }
  if (hi == 0) {
    const int n = n0 + lr;
    f16* op = out + ((size_t)(b * H_ + h) * W_ + n) * 8;
#pragma unroll
    for (int r = 0; r < 8; ++r) op[r] = (f16)(acc[r] + bias[r]);
  }
}

// v projection: O=64; store v_nchw [b][c][h][w] and v_wch [b][w][c][h]
__global__ __launch_bounds__(256) void k_proj_v(const f16* __restrict__ xin,
                                                const f16* __restrict__ w16,
                                                const float* __restrict__ bias,
                                                f16* __restrict__ vn,
                                                f16* __restrict__ vw) {
  const int bh = blockIdx.x;
  const int b = bh >> 7, h = bh & 127;
  const int wv = threadIdx.x >> 5, lane = threadIdx.x & 31;
  const int lr = lane & 15, hi = lane >> 4;
  const int m0 = (wv & 3) * 16;
  const int n0 = (wv >> 2) * 64;

  v8f acc[4];
#pragma unroll
  for (int j = 0; j < 4; ++j) acc[j] = acc_zero();
  const f16* xrow = xin + ((size_t)(b * H2 + h + 1) * W2 + 1) * IN_;
#pragma unroll
  for (int c0 = 0; c0 < IN_; c0 += 32) {
    v16h a = frag_load(w16 + (m0 + lr) * IN_ + c0, hi);
    v16h bfr[4];
#pragma unroll
    for (int j = 0; j < 4; ++j)
      bfr[j] = frag_load(xrow + (size_t)(n0 + j * 16 + lr) * IN_ + c0, hi);
#pragma unroll
    for (int j = 0; j < 4; ++j)
      acc[j] = WMMA(a, bfr[j], acc[j]);
  }
#pragma unroll
  for (int j = 0; j < 4; ++j) {
    const int n = n0 + j * 16 + lr;  // w coordinate
#pragma unroll
    for (int r = 0; r < 8; ++r) {
      const int c = m0 + hi * 8 + r;
      f16 val = (f16)(acc[j][r] + bias[c]);
      vn[((size_t)(b * IN_ + c) * H_ + h) * W_ + n] = val;
      vw[((size_t)(b * W_ + n) * IN_ + c) * H_ + h] = val;
    }
  }
}

// eH logits: per (b,w) column; e16[b][i][w][j], j<128, diag -> -30000
__global__ __launch_bounds__(256) void k_logits_h(const f16* __restrict__ q,
                                                  const f16* __restrict__ k,
                                                  f16* __restrict__ e) {
  __shared__ v8h qc[128];
  __shared__ v8h kc[128];
  const int b = blockIdx.x >> 7, w = blockIdx.x & 127;
  const int t = threadIdx.x;
  if (t < 128)
    qc[t] = *reinterpret_cast<const v8h*>(q + ((size_t)(b * H_ + t) * W_ + w) * 8);
  else
    kc[t - 128] = *reinterpret_cast<const v8h*>(k + ((size_t)(b * H_ + (t - 128)) * W_ + w) * 8);
  __syncthreads();
  const int i = t >> 1;
  const int j0 = (t & 1) * 64;
  v8h qi = qc[i];
  f16* er = e + ((size_t)(b * H_ + i) * W_ + w) * 256;
  for (int jj = 0; jj < 64; ++jj) {
    int j = j0 + jj;
    float v = dot8(qi, kc[j]);
    if (i == j) v = -30000.f;
    er[j] = (f16)v;
  }
}

// eW logits: per (b,h) row; e16[b][h][i][128+j]
__global__ __launch_bounds__(256) void k_logits_w(const f16* __restrict__ q,
                                                  const f16* __restrict__ k,
                                                  f16* __restrict__ e) {
  __shared__ v8h qc[128];
  __shared__ v8h kc[128];
  const int b = blockIdx.x >> 7, h = blockIdx.x & 127;
  const int t = threadIdx.x;
  if (t < 128)
    qc[t] = *reinterpret_cast<const v8h*>(q + ((size_t)(b * H_ + h) * W_ + t) * 8);
  else
    kc[t - 128] = *reinterpret_cast<const v8h*>(k + ((size_t)(b * H_ + h) * W_ + (t - 128)) * 8);
  __syncthreads();
  const int i = t >> 1;
  const int j0 = (t & 1) * 64;
  v8h qi = qc[i];
  f16* er = e + ((size_t)(b * H_ + h) * W_ + i) * 256 + 128;
  for (int jj = 0; jj < 64; ++jj) {
    int j = j0 + jj;
    er[j] = (f16)dot8(qi, kc[j]);
  }
}

// softmax over 256 logits per pixel, in place; one wave per pixel
__global__ __launch_bounds__(256) void k_softmax(f16* __restrict__ e) {
  const int wv = threadIdx.x >> 5, lane = threadIdx.x & 31;
  const size_t pix = (size_t)blockIdx.x * 8 + wv;
  v8h* row = reinterpret_cast<v8h*>(e + pix * 256);
  v8h d = row[lane];
  float f[8];
  float m = -1e30f;
#pragma unroll
  for (int i = 0; i < 8; ++i) { f[i] = (float)d[i]; m = fmaxf(m, f[i]); }
#pragma unroll
  for (int off = 16; off > 0; off >>= 1) m = fmaxf(m, __shfl_xor(m, off));
  float s = 0.f;
#pragma unroll
  for (int i = 0; i < 8; ++i) { f[i] = __expf(f[i] - m); s += f[i]; }
#pragma unroll
  for (int off = 16; off > 0; off >>= 1) s += __shfl_xor(s, off);
  const float inv = 1.f / s;
  v8h o;
#pragma unroll
  for (int i = 0; i < 8; ++i) o[i] = (f16)(f[i] * inv);
  row[lane] = o;
}

// outW: per (b,h): s[w][c] = 2 + sum_j attW[w][j] * v_nchw[b][c][h][j]
__global__ __launch_bounds__(256) void k_outw(const f16* __restrict__ e,
                                              const f16* __restrict__ vn,
                                              float* __restrict__ sbuf) {
  const int b = blockIdx.x >> 7, h = blockIdx.x & 127;
  const int wv = threadIdx.x >> 5, lane = threadIdx.x & 31;
  const int lr = lane & 15, hi = lane >> 4;
  const int m0 = wv * 16;  // w strip

  v8f acc[4];
#pragma unroll
  for (int j = 0; j < 4; ++j) acc[j] = acc_zero();

  const f16* arow = e + ((size_t)(b * H_ + h) * W_ + (m0 + lr)) * 256 + 128;
#pragma unroll
  for (int k0 = 0; k0 < 128; k0 += 32) {
    v16h a = frag_load(arow + k0, hi);
    v16h bfr[4];
#pragma unroll
    for (int jt = 0; jt < 4; ++jt)
      bfr[jt] = frag_load(vn + ((size_t)(b * IN_ + jt * 16 + lr) * H_ + h) * W_ + k0, hi);
#pragma unroll
    for (int jt = 0; jt < 4; ++jt)
      acc[jt] = WMMA(a, bfr[jt], acc[jt]);
  }
#pragma unroll
  for (int jt = 0; jt < 4; ++jt) {
    const int c = jt * 16 + lr;
#pragma unroll
    for (int r = 0; r < 8; ++r) {
      const int w = m0 + hi * 8 + r;
      sbuf[((size_t)(b * H_ + h) * W_ + w) * IN_ + c] = acc[jt][r] + 2.0f;
    }
  }
}

// outH: per (b,w): out[c][h] = sum_j v_wch[b][w][c][j] * attH[b][h][w][j];
// then xe += g1*(sbuf+outH), xq += g2*(sbuf+outH)   (in place, padded f16)
__global__ __launch_bounds__(256) void k_outh(const f16* __restrict__ e,
                                              const f16* __restrict__ vw,
                                              const float* __restrict__ sbuf,
                                              f16* __restrict__ xe,
                                              f16* __restrict__ xq,
                                              const float* __restrict__ g1p,
                                              const float* __restrict__ g2p) {
  const int b = blockIdx.x >> 7, w = blockIdx.x & 127;
  const int wv = threadIdx.x >> 5, lane = threadIdx.x & 31;
  const int lr = lane & 15, hi = lane >> 4;
  const int m0 = (wv & 3) * 16;   // c strip
  const int n0 = (wv >> 2) * 64;  // h half

  v8f acc[4];
#pragma unroll
  for (int j = 0; j < 4; ++j) acc[j] = acc_zero();

  const f16* arow = vw + ((size_t)(b * W_ + w) * IN_ + (m0 + lr)) * H_;
#pragma unroll
  for (int k0 = 0; k0 < 128; k0 += 32) {
    v16h a = frag_load(arow + k0, hi);
    v16h bfr[4];
#pragma unroll
    for (int j = 0; j < 4; ++j)
      bfr[j] = frag_load(e + ((size_t)(b * H_ + n0 + j * 16 + lr) * W_ + w) * 256 + k0, hi);
#pragma unroll
    for (int j = 0; j < 4; ++j)
      acc[j] = WMMA(a, bfr[j], acc[j]);
  }
  const float g1 = *g1p, g2 = *g2p;
#pragma unroll
  for (int j = 0; j < 4; ++j) {
    const int hh = n0 + j * 16 + lr;
#pragma unroll
    for (int r = 0; r < 8; ++r) {
      const int c = m0 + hi * 8 + r;
      const size_t sidx = ((size_t)(b * H_ + hh) * W_ + w) * IN_ + c;
      const size_t xidx = ((size_t)(b * H2 + hh + 1) * W2 + (w + 1)) * IN_ + c;
      const float s = sbuf[sidx] + acc[j][r];
      xe[xidx] = (f16)((float)xe[xidx] + g1 * s);
      xq[xidx] = (f16)((float)xq[xidx] + g2 * s);
    }
  }
}

// ---------------- host side ----------------

extern "C" void kernel_launch(void* const* d_in, const int* in_sizes, int n_in,
                              void* d_out, int out_size, void* d_ws, size_t ws_size,
                              hipStream_t stream) {
  (void)in_sizes; (void)n_in; (void)out_size; (void)ws_size;
  const float* x_ex   = (const float*)d_in[0];
  const float* x_q    = (const float*)d_in[1];
  const float* cw_ex  = (const float*)d_in[2];
  const float* bns_ex = (const float*)d_in[3];
  const float* bnb_ex = (const float*)d_in[4];
  const float* cw_q   = (const float*)d_in[5];
  const float* bns_q  = (const float*)d_in[6];
  const float* bnb_q  = (const float*)d_in[7];
  const float* q_w    = (const float*)d_in[8];
  const float* q_b    = (const float*)d_in[9];
  const float* k_w    = (const float*)d_in[10];
  const float* k_b    = (const float*)d_in[11];
  const float* v_w    = (const float*)d_in[12];
  const float* v_b    = (const float*)d_in[13];
  const float* g1     = (const float*)d_in[14];
  const float* g2     = (const float*)d_in[15];
  const float* tw_ex  = (const float*)d_in[16];
  const float* bns_tx = (const float*)d_in[17];
  const float* bnb_tx = (const float*)d_in[18];
  const float* tw_q   = (const float*)d_in[19];
  const float* bns_tq = (const float*)d_in[20];
  const float* bnb_tq = (const float*)d_in[21];
  float* out = (float*)d_out;

  // workspace carve-up (256B aligned chunks)
  char* p = (char*)d_ws;
  auto take = [&](size_t bytes) {
    char* r = p;
    p += (bytes + 255) & ~(size_t)255;
    return r;
  };
  const size_t npadC  = (size_t)B_ * H2 * W2 * C_;   // padded 256-ch image
  const size_t npadI  = (size_t)B_ * H2 * W2 * IN_;  // padded 64-ch image
  f16* xex16 = (f16*)take(npadC * 2);
  f16* xqn16 = (f16*)take(npadC * 2);
  f16* wex   = (f16*)take((size_t)64 * 9 * C_ * 2);
  f16* wq    = (f16*)take((size_t)64 * 9 * C_ * 2);
  f16* wtex  = (f16*)take((size_t)C_ * 9 * IN_ * 2);
  f16* wtq   = (f16*)take((size_t)C_ * 9 * IN_ * 2);
  f16* wq16  = (f16*)take((size_t)16 * 64 * 2);
  f16* wk16  = (f16*)take((size_t)16 * 64 * 2);
  f16* wv16  = (f16*)take((size_t)64 * 64 * 2);
  f16* xe16  = (f16*)take(npadI * 2);
  f16* xq16  = (f16*)take(npadI * 2);
  f16* q16   = (f16*)take((size_t)B_ * H_ * W_ * 8 * 2);
  f16* k16   = (f16*)take((size_t)B_ * H_ * W_ * 8 * 2);
  f16* vn    = (f16*)take((size_t)B_ * H_ * W_ * IN_ * 2);
  f16* vw    = (f16*)take((size_t)B_ * H_ * W_ * IN_ * 2);
  f16* e16   = (f16*)take((size_t)B_ * H_ * W_ * 256 * 2);
  float* sb  = (float*)take((size_t)B_ * H_ * W_ * IN_ * 4);

  const int BH = B_ * H_;  // 1024
  const int BW = B_ * W_;  // 1024

  // 1) layout conversions (padded inputs, zeroed intermediates)
  k_cvt_pad<<<(int)((npadC + 255) / 256), 256, 0, stream>>>(x_ex, xex16, (int)npadC);
  k_cvt_pad<<<(int)((npadC + 255) / 256), 256, 0, stream>>>(x_q, xqn16, (int)npadC);
  k_zero16<<<(int)((npadI + 255) / 256), 256, 0, stream>>>(xe16, (int)npadI);
  k_zero16<<<(int)((npadI + 255) / 256), 256, 0, stream>>>(xq16, (int)npadI);
  {
    int tot = 64 * 9 * C_;  // 147456
    k_reorder_w<<<(tot + 255) / 256, 256, 0, stream>>>(cw_ex, wex, C_, tot);
    k_reorder_w<<<(tot + 255) / 256, 256, 0, stream>>>(cw_q, wq, C_, tot);
  }
  {
    int tot = C_ * 9 * IN_;  // 147456
    k_reorder_w<<<(tot + 255) / 256, 256, 0, stream>>>(tw_ex, wtex, IN_, tot);
    k_reorder_w<<<(tot + 255) / 256, 256, 0, stream>>>(tw_q, wtq, IN_, tot);
  }
  k_pad_qk<<<4, 256, 0, stream>>>(q_w, wq16);
  k_pad_qk<<<4, 256, 0, stream>>>(k_w, wk16);
  k_cvt_small<<<16, 256, 0, stream>>>(v_w, wv16, 64 * 64);

  // 2) head convs
  k_conv_head<<<BH, 256, 0, stream>>>(xex16, wex, bns_ex, bnb_ex, xe16);
  k_conv_head<<<BH, 256, 0, stream>>>(xqn16, wq, bns_q, bnb_q, xq16);

  // 3) two criss-cross attention iterations (in place on xe16/xq16)
  for (int it = 0; it < 2; ++it) {
    k_proj_qk<<<BH, 256, 0, stream>>>(xq16, wq16, q_b, q16);
    k_proj_qk<<<BH, 256, 0, stream>>>(xe16, wk16, k_b, k16);
    k_proj_v<<<BH, 256, 0, stream>>>(xe16, wv16, v_b, vn, vw);
    k_logits_h<<<BW, 256, 0, stream>>>(q16, k16, e16);
    k_logits_w<<<BH, 256, 0, stream>>>(q16, k16, e16);
    k_softmax<<<(B_ * H_ * W_) / 8, 256, 0, stream>>>(e16);
    k_outw<<<BH, 256, 0, stream>>>(e16, vn, sb);
    k_outh<<<BW, 256, 0, stream>>>(e16, vw, sb, xe16, xq16, g1, g2);
  }

  // 4) tail convs -> f32 NCHW outputs
  const size_t half = (size_t)B_ * C_ * H_ * W_;
  k_conv_tail<<<dim3(BH, 4), 256, 0, stream>>>(xe16, wtex, bns_tx, bnb_tx, out);
  k_conv_tail<<<dim3(BH, 4), 256, 0, stream>>>(xq16, wtq, bns_tq, bnb_tq, out + half);
}
